// GptOssAttention_78752520339570
// MI455X (gfx1250) — matrix operand order repaired and checked by reading, not compiled, not run on hardware
//
#include <hip/hip_runtime.h>
#include <hip/hip_bf16.h>
#include <math.h>

typedef _Float16 h4   __attribute__((ext_vector_type(4)));
typedef _Float16 h8   __attribute__((ext_vector_type(8)));
typedef _Float16 v16h __attribute__((ext_vector_type(16)));
typedef float    v8f  __attribute__((ext_vector_type(8)));
typedef int      gv4i __attribute__((vector_size(16)));   // matches builtin param

#define N_TOK 1024
#define HDIM  2880
#define NHq   64
#define NKVh  8
#define DHEAD 64
#define QKVD  5120   /* (64 + 2*8) * 64 */
#define WWIN  128

__device__ __forceinline__ v16h cat8(h8 a, h8 b) {
  return __builtin_shufflevector(a, b, 0,1,2,3,4,5,6,7,8,9,10,11,12,13,14,15);
}
__device__ __forceinline__ v8f wmma16(v16h a, v16h b, v8f c) {
  return __builtin_amdgcn_wmma_f32_16x16x32_f16(false, a, false, b, (short)0, c,
                                                false, false);
}
__device__ __forceinline__ v8f zero8() {
  v8f z = {0.f,0.f,0.f,0.f,0.f,0.f,0.f,0.f};
  return z;
}

// 16-byte global -> LDS copy on the CDNA5 async path (ASYNCcnt, no VGPR hop).
__device__ __forceinline__ void cp16_async(void* lds, const void* g) {
#if __has_builtin(__builtin_amdgcn_global_load_async_to_lds_b128)
  __builtin_amdgcn_global_load_async_to_lds_b128(
      (__attribute__((address_space(1))) gv4i*)g,
      (__attribute__((address_space(3))) gv4i*)lds, 0, 0);
#else
  *(uint4*)lds = *(const uint4*)g;
#endif
}
__device__ __forceinline__ void wait_async0() {
#if __has_builtin(__builtin_amdgcn_s_wait_asynccnt)
  __builtin_amdgcn_s_wait_asynccnt(0);
#elif defined(__HIP_DEVICE_COMPILE__)
  asm volatile("s_wait_asynccnt 0" ::: "memory");
#endif
}

union U4H8 { uint4 u; h8 h; };

// ---------------------------------------------------------------- f32 -> f16
__global__ __launch_bounds__(256) void cvt_f32_f16(const float* __restrict__ in,
                                                   _Float16* __restrict__ out,
                                                   int n4) {
  int i = blockIdx.x * blockDim.x + threadIdx.x;
  int stride = gridDim.x * blockDim.x;
  for (; i < n4; i += stride) {
    float4 v = ((const float4*)in)[i];
    h4 o;
    o[0] = (_Float16)v.x; o[1] = (_Float16)v.y;
    o[2] = (_Float16)v.z; o[3] = (_Float16)v.w;
    ((h4*)out)[i] = o;
  }
}

// ------------------------------------------------- WMMA GEMM  C = A(f16)*B(f16)
// A: M x K row-major (M % 128 == 0, K % 32 == 0), B: K x Ncols row-major.
// Block = 128x128 tile of C, 8 waves (4 M x 2 N), each wave 32x64 (2x4 frags).
// Double-buffered LDS; A tile staged via async global->LDS b128.
__global__ __launch_bounds__(256) void gemm_f16_wmma(
    const _Float16* __restrict__ A, const _Float16* __restrict__ B,
    float* __restrict__ C, int K, int Ncols) {
  __shared__ __attribute__((aligned(16))) _Float16 Ash[2][128][40]; // [m][k]
  __shared__ __attribute__((aligned(16))) _Float16 Bsh[2][128][40]; // [n][k] (T)

  const int tid  = threadIdx.x;
  const int lane = tid & 31;
  const int wave = tid >> 5;
  const int wm   = wave >> 1;      // 0..3
  const int wn   = wave & 1;       // 0..1
  const int hi   = lane >> 4;      // lane half
  const int lo16 = lane & 15;
  const int blockM = blockIdx.y * 128;
  const int blockN = blockIdx.x * 128;
  const int KT = K >> 5;

  auto stageA = [&](int buf, int k0) {
#pragma unroll
    for (int i = 0; i < 2; ++i) {
      int c = tid + 256 * i;           // 512 chunks of 8 halfs
      int row = c >> 2, kc = (c & 3) * 8;
      cp16_async(&Ash[buf][row][kc],
                 A + (size_t)(blockM + row) * K + k0 + kc);
    }
  };
  auto stageB = [&](int buf, int k0) {
    uint4 bv[2];
#pragma unroll
    for (int i = 0; i < 2; ++i) {
      int c = tid + 256 * i;
      int kr = c >> 4, nc = (c & 15) * 8;
      int gn = blockN + nc;
      if (gn < Ncols) {
        bv[i] = *(const uint4*)(B + (size_t)(k0 + kr) * Ncols + gn);
      } else {
        bv[i].x = 0u; bv[i].y = 0u; bv[i].z = 0u; bv[i].w = 0u;
      }
    }
#pragma unroll
    for (int i = 0; i < 2; ++i) {
      int c = tid + 256 * i;
      int kr = c >> 4, nc = (c & 15) * 8;
      U4H8 u; u.u = bv[i];
#pragma unroll
      for (int j = 0; j < 8; ++j) Bsh[buf][nc + j][kr] = u.h[j];
    }
  };

  v8f acc[2][4];
#pragma unroll
  for (int mf = 0; mf < 2; ++mf)
#pragma unroll
    for (int nf = 0; nf < 4; ++nf) acc[mf][nf] = zero8();

  stageA(0, 0);
  stageB(0, 0);
  wait_async0();
  __syncthreads();

  for (int kt = 0; kt < KT; ++kt) {
    const int buf = kt & 1;
    if (kt + 1 < KT) {               // prefetch next tile while computing
      stageA(buf ^ 1, (kt + 1) * 32);
      stageB(buf ^ 1, (kt + 1) * 32);
    }

    v16h af[2], bf[4];
#pragma unroll
    for (int mf = 0; mf < 2; ++mf) {
      int row = wm * 32 + mf * 16 + lo16;
      h8 l = *(const h8*)&Ash[buf][row][hi * 8];
      h8 h = *(const h8*)&Ash[buf][row][16 + hi * 8];
      af[mf] = cat8(l, h);
    }
#pragma unroll
    for (int nf = 0; nf < 4; ++nf) {
      int row = wn * 64 + nf * 16 + lo16;
      h8 l = *(const h8*)&Bsh[buf][row][hi * 8];
      h8 h = *(const h8*)&Bsh[buf][row][16 + hi * 8];
      bf[nf] = cat8(l, h);
    }
#pragma unroll
    for (int mf = 0; mf < 2; ++mf)
#pragma unroll
      for (int nf = 0; nf < 4; ++nf)
        acc[mf][nf] = wmma16(af[mf], bf[nf], acc[mf][nf]);

    wait_async0();
    __syncthreads();
  }

#pragma unroll
  for (int mf = 0; mf < 2; ++mf)
#pragma unroll
    for (int nf = 0; nf < 4; ++nf) {
      int col = blockN + wn * 64 + nf * 16 + lo16;
      if (col < Ncols) {
#pragma unroll
        for (int r = 0; r < 8; ++r) {
          int row = blockM + wm * 32 + mf * 16 + r + 8 * hi;
          C[(size_t)row * Ncols + col] = acc[mf][nf][r];
        }
      }
    }
}

// ------------------------------------------------------------- RoPE + pack f16
__global__ __launch_bounds__(256) void rope_pack(
    const float* __restrict__ qkv, const int* __restrict__ positions,
    _Float16* __restrict__ qh, _Float16* __restrict__ kh,
    _Float16* __restrict__ vh) {
  const int n = blockIdx.x;
  const int t = threadIdx.x;
  const float pos = (float)positions[n];
  const float* row = qkv + (size_t)n * QKVD;
  const float LOG_THETA = 11.918390573078392f;  // ln(150000)

  for (int idx = t; idx < 2048; idx += 256) {    // q: 64 heads * 32 pairs
    int h = idx >> 5, j = idx & 31;
    float freq = __expf(-(float)j * (1.0f / 32.0f) * LOG_THETA);
    float s, c;
    __sincosf(pos * freq, &s, &c);
    float x1 = row[h * 64 + j], x2 = row[h * 64 + j + 32];
    size_t base = ((size_t)n * NHq + h) * DHEAD;
    qh[base + j]      = (_Float16)(x1 * c - x2 * s);
    qh[base + j + 32] = (_Float16)(x2 * c + x1 * s);
  }
  for (int idx = t; idx < 256; idx += 256) {     // k: 8 heads * 32 pairs
    int h = idx >> 5, j = idx & 31;
    float freq = __expf(-(float)j * (1.0f / 32.0f) * LOG_THETA);
    float s, c;
    __sincosf(pos * freq, &s, &c);
    float x1 = row[4096 + h * 64 + j], x2 = row[4096 + h * 64 + j + 32];
    size_t base = ((size_t)n * NKVh + h) * DHEAD;
    kh[base + j]      = (_Float16)(x1 * c - x2 * s);
    kh[base + j + 32] = (_Float16)(x2 * c + x1 * s);
  }
  for (int idx = t; idx < 512; idx += 256) {     // v: plain converts
    vh[(size_t)n * 512 + idx] = (_Float16)row[4608 + idx];
  }
}

// ----------------------------------------------------- sliding-window attention
// grid = (N/16 token blocks, NKV kv-heads); 8 waves = 8 grouped q-heads.
__global__ __launch_bounds__(256) void attn_window(
    const _Float16* __restrict__ qh, const _Float16* __restrict__ kh,
    const _Float16* __restrict__ vh, const float* __restrict__ sinks,
    _Float16* __restrict__ attn) {
  __shared__ __attribute__((aligned(16))) _Float16 Ksh[144][64];   // [key][d]
  __shared__ __attribute__((aligned(16))) _Float16 Vt[64][160];    // [d][key]
  __shared__ __attribute__((aligned(16))) _Float16 Pw[8][16][32];  // per-wave P

  const int tid  = threadIdx.x;
  const int lane = tid & 31;
  const int wave = tid >> 5;
  const int hi   = lane >> 4;
  const int lo16 = lane & 15;
  const int qb   = blockIdx.x * 16;
  const int kv   = blockIdx.y;
  const int head = kv * 8 + wave;
  const int kbase = qb - 128;  // 9 tiles of 16 cover [qb-128, qb+15]

  // ---- K stage via async global->LDS: 144 rows * 64 halfs = 1152 chunks of 8
  for (int c = tid; c < 1152; c += 256) {
    int p = c >> 3, off = (c & 7) * 8;
    int tok = kbase + p;
    tok = tok < 0 ? 0 : (tok > N_TOK - 1 ? N_TOK - 1 : tok);
    cp16_async(&Ksh[p][off], kh + ((size_t)tok * NKVh + kv) * DHEAD + off);
  }
  // ---- V transposed stage (+ zero the 16-key pad)
  for (int p = tid; p < 160; p += 256) {
    if (p < 144) {
      int tok = kbase + p;
      tok = tok < 0 ? 0 : (tok > N_TOK - 1 ? N_TOK - 1 : tok);
      const _Float16* src = vh + ((size_t)tok * NKVh + kv) * DHEAD;
#pragma unroll
      for (int d = 0; d < 64; ++d) Vt[d][p] = src[d];
    } else {
#pragma unroll
      for (int d = 0; d < 64; ++d) Vt[d][p] = (_Float16)0.0f;
    }
  }
  wait_async0();
  __syncthreads();

  // ---- Q fragments straight from global (A layout, two 32-wide d chunks)
  const int qtokL = qb + lo16;
  v16h qf[2];
#pragma unroll
  for (int c = 0; c < 2; ++c) {
    const _Float16* src = qh + ((size_t)qtokL * NHq + head) * DHEAD + c * 32;
    h8 l = *(const h8*)(src + hi * 8);
    h8 h = *(const h8*)(src + 16 + hi * 8);
    qf[c] = cat8(l, h);
  }

  // ---- scores: 9 tiles of 16 keys, S = Q * K^T (f32 accum)
  v8f S[9];
#pragma unroll
  for (int t = 0; t < 9; ++t) {
    v8f cc = zero8();
    int krow = t * 16 + lo16;
#pragma unroll
    for (int c = 0; c < 2; ++c) {
      h8 l = *(const h8*)&Ksh[krow][c * 32 + hi * 8];
      h8 h = *(const h8*)&Ksh[krow][c * 32 + 16 + hi * 8];
      cc = wmma16(qf[c], cat8(l, h), cc);
    }
    S[t] = cc;
  }

  // ---- masked softmax with sink (rows = queries, 16-lane group reductions)
  const float sink = sinks[head];
  const float scale = 0.125f;  // D^-0.5
  float denom[8];
#pragma unroll
  for (int r = 0; r < 8; ++r) {
    int qt = qb + r + 8 * hi;
    float mx = sink;
#pragma unroll
    for (int t = 0; t < 9; ++t) {
      int key = kbase + t * 16 + lo16;
      bool valid = (key >= 0) && (key >= qt - (WWIN - 1)) && (key <= qt);
      float s = valid ? S[t][r] * scale : -__builtin_inff();
      S[t][r] = s;
      mx = fmaxf(mx, s);
    }
#pragma unroll
    for (int m = 1; m < 16; m <<= 1) mx = fmaxf(mx, __shfl_xor(mx, m, 32));
    float sum = 0.f;
#pragma unroll
    for (int t = 0; t < 9; ++t) {
      float p = __expf(S[t][r] - mx);
      S[t][r] = p;
      sum += p;
    }
#pragma unroll
    for (int m = 1; m < 16; m <<= 1) sum += __shfl_xor(sum, m, 32);
    denom[r] = sum + __expf(sink - mx);
  }

  // ---- O = P * V via paired key tiles (A = 16x32 P slice staged in LDS)
  v8f O[4];
#pragma unroll
  for (int nt = 0; nt < 4; ++nt) O[nt] = zero8();

#pragma unroll
  for (int pr = 0; pr < 5; ++pr) {
#pragma unroll
    for (int half = 0; half < 2; ++half) {
      int t = pr * 2 + half;
#pragma unroll
      for (int r = 0; r < 8; ++r) {
        _Float16 val = (t < 9) ? (_Float16)S[t][r] : (_Float16)0.0f;
        Pw[wave][r + 8 * hi][half * 16 + lo16] = val;
      }
    }
    // per-wave LDS store->load ordering (CDNA5 split dependency counter)
    asm volatile("s_wait_dscnt 0" ::: "memory");

    h8 pl = *(const h8*)&Pw[wave][lo16][hi * 8];
    h8 ph = *(const h8*)&Pw[wave][lo16][16 + hi * 8];
    v16h pf = cat8(pl, ph);
#pragma unroll
    for (int nt = 0; nt < 4; ++nt) {
      int d = nt * 16 + lo16;
      h8 bl = *(const h8*)&Vt[d][pr * 32 + hi * 8];
      h8 bh = *(const h8*)&Vt[d][pr * 32 + 16 + hi * 8];
      O[nt] = wmma16(pf, cat8(bl, bh), O[nt]);
    }
    asm volatile("s_wait_dscnt 0" ::: "memory");
  }

  // ---- normalize and write f16 attn output [tok][head*64 + d]
#pragma unroll
  for (int nt = 0; nt < 4; ++nt) {
    int d = nt * 16 + lo16;
#pragma unroll
    for (int r = 0; r < 8; ++r) {
      int tok = qb + r + 8 * hi;
      attn[((size_t)tok * NHq + head) * DHEAD + d] =
          (_Float16)(O[nt][r] / denom[r]);
    }
  }
}

// ------------------------------------------------------------------- launcher
extern "C" void kernel_launch(void* const* d_in, const int* in_sizes, int n_in,
                              void* d_out, int out_size, void* d_ws,
                              size_t ws_size, hipStream_t stream) {
  const float* hidden = (const float*)d_in[0];   // [1024][2880]
  const float* Wqkv   = (const float*)d_in[1];   // [2880][5120]
  const float* Wo     = (const float*)d_in[2];   // [4096][2880]
  const float* sinks  = (const float*)d_in[3];   // [64]
  const int*   pos    = (const int*)d_in[4];     // [1024]
  float* out = (float*)d_out;                    // [1024][2880]

  char* ws = (char*)d_ws;
  size_t off = 0;
  _Float16* hidden_h = (_Float16*)(ws + off); off += (size_t)N_TOK * HDIM * 2;
  _Float16* Wqkv_h   = (_Float16*)(ws + off); off += (size_t)HDIM * QKVD * 2;
  _Float16* Wo_h     = (_Float16*)(ws + off); off += (size_t)4096 * HDIM * 2;
  float*    qkv_f    = (float*)(ws + off);    off += (size_t)N_TOK * QKVD * 4;
  _Float16* q_h      = (_Float16*)(ws + off); off += (size_t)N_TOK * 4096 * 2;
  _Float16* k_h      = (_Float16*)(ws + off); off += (size_t)N_TOK * 512 * 2;
  _Float16* v_h      = (_Float16*)(ws + off); off += (size_t)N_TOK * 512 * 2;
  _Float16* attn_h   = (_Float16*)(ws + off); off += (size_t)N_TOK * 4096 * 2;

  // 1) convert inputs to f16
  cvt_f32_f16<<<2880, 256, 0, stream>>>(hidden, hidden_h, N_TOK * HDIM / 4);
  cvt_f32_f16<<<14400, 256, 0, stream>>>(Wqkv, Wqkv_h, HDIM * QKVD / 4);
  cvt_f32_f16<<<11520, 256, 0, stream>>>(Wo, Wo_h, 4096 * HDIM / 4);

  // 2) QKV projection: [1024 x 2880] * [2880 x 5120] -> f32
  gemm_f16_wmma<<<dim3(QKVD / 128, N_TOK / 128), 256, 0, stream>>>(
      hidden_h, Wqkv_h, qkv_f, HDIM, QKVD);

  // 3) RoPE + pack to f16 q/k/v
  rope_pack<<<N_TOK, 256, 0, stream>>>(qkv_f, pos, q_h, k_h, v_h);

  // 4) sliding-window attention with sinks
  attn_window<<<dim3(N_TOK / 16, NKVh), 256, 0, stream>>>(q_h, k_h, v_h, sinks,
                                                          attn_h);

  // 5) output projection: [1024 x 4096] * [4096 x 2880] -> f32 out
  gemm_f16_wmma<<<dim3((HDIM + 127) / 128, N_TOK / 128), 256, 0, stream>>>(
      attn_h, Wo_h, out, 4096, HDIM);
}
